// MixerBlock_31507880083766
// MI455X (gfx1250) — compile-verified, matrix operand order checked by reading, not run on hardware
//
#include <hip/hip_runtime.h>
#include <cstdint>
#include <cstddef>

typedef __attribute__((ext_vector_type(16))) __bf16 v16bf;
typedef __attribute__((ext_vector_type(8)))  float  v8f;

union Frag {
  v16bf bf;
  uint4 q[2];
};

__device__ __forceinline__ unsigned short f2bf(float f){
  union { float f; unsigned u; } v; v.f = f;
  unsigned r = v.u + 0x7FFFu + ((v.u >> 16) & 1u);
  return (unsigned short)(r >> 16);
}
__device__ __forceinline__ float bf2f(unsigned short h){
  union { unsigned u; float f; } v; v.u = ((unsigned)h) << 16; return v.f;
}
__device__ __forceinline__ unsigned pack2(float a, float b){
  return (unsigned)f2bf(a) | ((unsigned)f2bf(b) << 16);
}
__device__ __forceinline__ v8f vzero(){
  v8f z;
#pragma unroll
  for (int i = 0; i < 8; ++i) z[i] = 0.f;
  return z;
}
__device__ __forceinline__ v8f wmma_bf16(const Frag& a, const Frag& b, v8f c){
  return __builtin_amdgcn_wmma_f32_16x16x32_bf16(false, a.bf, false, b.bf, (short)0, c, false, false);
}
// A-fragment (16x32 bf16, row m per lane&15, K-halves split by lane>>4)
__device__ __forceinline__ void load_a(Frag& f, const unsigned* A, int strideU, int m, int kt, int lane){
  const unsigned* p = A + (size_t)m * strideU + kt * 16 + ((lane >> 4) << 2);
  f.q[0] = *(const uint4*)p;        // K +0..7
  f.q[1] = *(const uint4*)(p + 8);  // K +16..23
}
// B-fragment from pre-swizzled weights: [kt][nt][lane][8 uints]
__device__ __forceinline__ void load_b(Frag& f, const unsigned* Bs, int NT, int kt, int nt, int lane){
  const unsigned* p = Bs + ((size_t)((kt * NT + nt) * 32 + lane) << 3);
  f.q[0] = *(const uint4*)p;
  f.q[1] = *(const uint4*)(p + 4);
}
__device__ __forceinline__ float siluf(float x){ return x / (1.f + __expf(-x)); }

// ---------------- prep kernels ----------------
__global__ void k_swz(const float* __restrict__ W, unsigned* __restrict__ dst, int K, int Ncols){
  int NT = Ncols >> 4;
  int total = (K * Ncols) >> 1;
  for (int f = blockIdx.x * blockDim.x + threadIdx.x; f < total; f += gridDim.x * blockDim.x){
    int u = f & 7, lane = (f >> 3) & 31, rest = f >> 8;
    int nt = rest % NT, kt = rest / NT;
    int n  = nt * 16 + (lane & 15);
    int kb = kt * 32 + ((lane >> 4) << 3) + ((u >> 2) << 4) + ((u & 3) << 1);
    dst[f] = pack2(W[(size_t)kb * Ncols + n], W[(size_t)(kb + 1) * Ncols + n]);
  }
}
__global__ void k_vin(const float* __restrict__ V, const float* __restrict__ S_,
                      unsigned* __restrict__ vin, int N){
  int total = N * 80;
  for (int f = blockIdx.x * blockDim.x + threadIdx.x; f < total; f += gridDim.x * blockDim.x){
    int n = f / 80, j = f - n * 80; int c0 = j << 1;
    float a, b;
    if (c0 < 128){ a = V[(size_t)n * 128 + c0]; b = V[(size_t)n * 128 + c0 + 1]; }
    else         { a = S_[(size_t)n * 32 + (c0 - 128)]; b = S_[(size_t)n * 32 + (c0 - 127)]; }
    vin[f] = pack2(a, b);
  }
}
__global__ void k_zero(float* __restrict__ p, long n){
  for (long i = (long)blockIdx.x * blockDim.x + threadIdx.x; i < n; i += (long)gridDim.x * blockDim.x)
    p[i] = 0.f;
}

// ---------------- edge kernel: e = LN(MLP(cat(E, Vin[snd], Vin[rcv]))) ----------------
__global__ __launch_bounds__(256) void k_edge(
  const float* __restrict__ E, const unsigned* __restrict__ vin,
  const int* __restrict__ snd, const int* __restrict__ rcv,
  const unsigned* __restrict__ We1s, const float* __restrict__ be1,
  const unsigned* __restrict__ We2s, const float* __restrict__ be2,
  const float* __restrict__ ge, const float* __restrict__ beL,
  float* __restrict__ outE, float* __restrict__ agg, int NE)
{
  __shared__ __align__(16) unsigned char smem[32 * 224 * 4 + 32 * 64 * 4];
  unsigned* A1 = (unsigned*)smem;                       // [32][224] bf16-pairs (K=448)
  unsigned* A2 = (unsigned*)(smem + 32 * 224 * 4);      // [32][64]  bf16-pairs (K=128)
  float*    Cb = (float*)smem;                          // [32][128] f32, overlays A1 (safe: A1 dead)
  __shared__ int sidx[32], ridx[32];
  __shared__ float red[32][8][2];

  int tid = threadIdx.x;
  int e0 = blockIdx.x * 32;
  if (tid < 32){
    int e = e0 + tid; if (e > NE - 1) e = NE - 1;
    sidx[tid] = snd[e]; ridx[tid] = rcv[e];
  }
  __syncthreads();
  for (int f = tid; f < 32 * 224; f += 256){
    int r = f / 224, u = f - r * 224;
    int e = e0 + r; if (e > NE - 1) e = NE - 1;
    unsigned val;
    if (u < 64){
      const float* ep = E + (size_t)e * 128 + (u << 1);
      val = pack2(ep[0], ep[1]);
    } else if (u < 144){
      val = vin[(size_t)sidx[r] * 80 + (u - 64)];
    } else {
      val = vin[(size_t)ridx[r] * 80 + (u - 144)];
    }
    A1[r * 224 + u] = val;
  }
  __syncthreads();

  int wave = tid >> 5, lane = tid & 31;
  int mt = wave & 1, ntb = (wave >> 1) << 1;            // 2 m-tiles x 8 n-tiles, 2 n-tiles/wave
  int m = mt * 16 + (lane & 15);
  v8f acc0 = vzero(), acc1 = vzero();
  for (int kt = 0; kt < 14; ++kt){
    Frag a;  load_a(a, A1, 224, m, kt, lane);
    Frag b0; load_b(b0, We1s, 8, kt, ntb, lane);
    Frag b1; load_b(b1, We1s, 8, kt, ntb + 1, lane);
    acc0 = wmma_bf16(a, b0, acc0);
    acc1 = wmma_bf16(a, b1, acc1);
  }
  {
    int hi = (lane >> 4) << 3;
#pragma unroll
    for (int i = 0; i < 8; ++i){
      int row = mt * 16 + i + hi;
      int c0 = ntb * 16 + (lane & 15), c1 = c0 + 16;
      ((unsigned short*)A2)[row * 128 + c0] = f2bf(siluf(acc0[i] + be1[c0]));
      ((unsigned short*)A2)[row * 128 + c1] = f2bf(siluf(acc1[i] + be1[c1]));
    }
  }
  __syncthreads();
  acc0 = vzero(); acc1 = vzero();
  for (int kt = 0; kt < 4; ++kt){
    Frag a;  load_a(a, A2, 64, m, kt, lane);
    Frag b0; load_b(b0, We2s, 8, kt, ntb, lane);
    Frag b1; load_b(b1, We2s, 8, kt, ntb + 1, lane);
    acc0 = wmma_bf16(a, b0, acc0);
    acc1 = wmma_bf16(a, b1, acc1);
  }
  {
    int hi = (lane >> 4) << 3;
#pragma unroll
    for (int i = 0; i < 8; ++i){
      int row = mt * 16 + i + hi;
      int c0 = ntb * 16 + (lane & 15), c1 = c0 + 16;
      Cb[row * 128 + c0] = acc0[i] + be2[c0];
      Cb[row * 128 + c1] = acc1[i] + be2[c1];
    }
  }
  __syncthreads();
  // fp32 LayerNorm + residual + scatter-add (8 threads per row)
  int r = tid >> 3, t8 = tid & 7;
  const float* crow = Cb + r * 128;
  float s = 0.f, s2 = 0.f;
  for (int c = t8 * 16; c < t8 * 16 + 16; ++c){ float x = crow[c]; s += x; s2 += x * x; }
  red[r][t8][0] = s; red[r][t8][1] = s2;
  __syncthreads();
  float sum = 0.f, sum2 = 0.f;
#pragma unroll
  for (int j = 0; j < 8; ++j){ sum += red[r][j][0]; sum2 += red[r][j][1]; }
  float mu = sum * (1.f / 128.f);
  float var = sum2 * (1.f / 128.f) - mu * mu;
  float rs = rsqrtf(var + 1e-5f);
  int e = e0 + r;
  if (e < NE){
    int rv = ridx[r];
    for (int c = t8 * 16; c < t8 * 16 + 16; ++c){
      float val = (crow[c] - mu) * rs * ge[c] + beL[c];
      outE[(size_t)e * 128 + c] = E[(size_t)e * 128 + c] + val;
      atomicAdd(&agg[(size_t)rv * 128 + c], val);
    }
  }
}

// ---------------- node kernel: v = LN(MLP(cat(Vin, agg))), outV = V + v ----------------
__global__ __launch_bounds__(256) void k_node(
  const float* __restrict__ V, const unsigned* __restrict__ vin, const float* __restrict__ agg,
  const unsigned* __restrict__ Wn1s, const float* __restrict__ bn1,
  const unsigned* __restrict__ Wn2s, const float* __restrict__ bn2,
  const float* __restrict__ gn, const float* __restrict__ bnL,
  float* __restrict__ outV, int N)
{
  __shared__ __align__(16) unsigned char smem[32 * 144 * 4 + 32 * 64 * 4];
  unsigned* A1 = (unsigned*)smem;                       // [32][144] (K=288)
  unsigned* A2 = (unsigned*)(smem + 32 * 144 * 4);      // [32][64]
  float*    Cb = (float*)smem;                          // [32][128] overlays A1
  __shared__ float red[32][8][2];

  int tid = threadIdx.x;
  int n0 = blockIdx.x * 32;
  for (int f = tid; f < 32 * 144; f += 256){
    int r = f / 144, u = f - r * 144;
    int n = n0 + r; if (n > N - 1) n = N - 1;
    unsigned val;
    if (u < 80) val = vin[(size_t)n * 80 + u];
    else {
      const float* ap = agg + (size_t)n * 128 + ((u - 80) << 1);
      val = pack2(ap[0], ap[1]);
    }
    A1[r * 144 + u] = val;
  }
  __syncthreads();

  int wave = tid >> 5, lane = tid & 31;
  int mt = wave & 1, ntb = (wave >> 1) << 1;
  int m = mt * 16 + (lane & 15);
  v8f acc0 = vzero(), acc1 = vzero();
  for (int kt = 0; kt < 9; ++kt){
    Frag a;  load_a(a, A1, 144, m, kt, lane);
    Frag b0; load_b(b0, Wn1s, 8, kt, ntb, lane);
    Frag b1; load_b(b1, Wn1s, 8, kt, ntb + 1, lane);
    acc0 = wmma_bf16(a, b0, acc0);
    acc1 = wmma_bf16(a, b1, acc1);
  }
  {
    int hi = (lane >> 4) << 3;
#pragma unroll
    for (int i = 0; i < 8; ++i){
      int row = mt * 16 + i + hi;
      int c0 = ntb * 16 + (lane & 15), c1 = c0 + 16;
      ((unsigned short*)A2)[row * 128 + c0] = f2bf(siluf(acc0[i] + bn1[c0]));
      ((unsigned short*)A2)[row * 128 + c1] = f2bf(siluf(acc1[i] + bn1[c1]));
    }
  }
  __syncthreads();
  acc0 = vzero(); acc1 = vzero();
  for (int kt = 0; kt < 4; ++kt){
    Frag a;  load_a(a, A2, 64, m, kt, lane);
    Frag b0; load_b(b0, Wn2s, 8, kt, ntb, lane);
    Frag b1; load_b(b1, Wn2s, 8, kt, ntb + 1, lane);
    acc0 = wmma_bf16(a, b0, acc0);
    acc1 = wmma_bf16(a, b1, acc1);
  }
  {
    int hi = (lane >> 4) << 3;
#pragma unroll
    for (int i = 0; i < 8; ++i){
      int row = mt * 16 + i + hi;
      int c0 = ntb * 16 + (lane & 15), c1 = c0 + 16;
      Cb[row * 128 + c0] = acc0[i] + bn2[c0];
      Cb[row * 128 + c1] = acc1[i] + bn2[c1];
    }
  }
  __syncthreads();
  int r = tid >> 3, t8 = tid & 7;
  const float* crow = Cb + r * 128;
  float s = 0.f, s2 = 0.f;
  for (int c = t8 * 16; c < t8 * 16 + 16; ++c){ float x = crow[c]; s += x; s2 += x * x; }
  red[r][t8][0] = s; red[r][t8][1] = s2;
  __syncthreads();
  float sum = 0.f, sum2 = 0.f;
#pragma unroll
  for (int j = 0; j < 8; ++j){ sum += red[r][j][0]; sum2 += red[r][j][1]; }
  float mu = sum * (1.f / 128.f);
  float rs = rsqrtf(sum2 * (1.f / 128.f) - mu * mu + 1e-5f);
  int n = n0 + r;
  if (n < N){
    for (int c = t8 * 16; c < t8 * 16 + 16; ++c){
      float val = (crow[c] - mu) * rs * gn[c] + bnL[c];
      outV[(size_t)n * 128 + c] = V[(size_t)n * 128 + c] + val;
    }
  }
}

// ---------------- LayerNorm -> bf16 activations ----------------
__global__ void k_ln(const float* __restrict__ X, const float* __restrict__ g, const float* __restrict__ b,
                     unsigned* __restrict__ out, int rows)
{
  for (int r = blockIdx.x * blockDim.x + threadIdx.x; r < rows; r += gridDim.x * blockDim.x){
    const float* x = X + (size_t)r * 128;
    float s = 0.f, s2 = 0.f;
    for (int c = 0; c < 128; ++c){ float t = x[c]; s += t; s2 += t * t; }
    float mu = s * (1.f / 128.f);
    float rs = rsqrtf(s2 * (1.f / 128.f) - mu * mu + 1e-5f);
    for (int j = 0; j < 64; ++j){
      int c = j << 1;
      out[(size_t)r * 64 + j] = pack2((x[c] - mu) * rs * g[c] + b[c],
                                      (x[c + 1] - mu) * rs * g[c + 1] + b[c + 1]);
    }
  }
}

// ---------------- attention: slice softmax weights ----------------
__global__ __launch_bounds__(256) void k_attslice(const unsigned* __restrict__ xln,
                                                  const float* __restrict__ Ws,
                                                  unsigned* __restrict__ wbuf, int N)
{
  __shared__ float WsL[32 * 64];
  int tid = threadIdx.x;
  for (int f = tid; f < 2048; f += 256) WsL[f] = Ws[f];
  __syncthreads();
  for (int n = blockIdx.x * 256 + tid; n < N; n += gridDim.x * 256){
    for (int h = 0; h < 4; ++h){
      float xv[32];
      const unsigned* xp = xln + (size_t)n * 64 + h * 16;
#pragma unroll
      for (int j = 0; j < 16; ++j){
        unsigned u = xp[j];
        xv[2 * j]     = bf2f((unsigned short)(u & 0xFFFFu));
        xv[2 * j + 1] = bf2f((unsigned short)(u >> 16));
      }
      float lg[64];
#pragma unroll
      for (int mm = 0; mm < 64; ++mm) lg[mm] = 0.f;
      for (int d = 0; d < 32; ++d){
        float xd = xv[d];
        const float* wr = WsL + d * 64;
#pragma unroll
        for (int mm = 0; mm < 64; ++mm) lg[mm] += xd * wr[mm];
      }
      float mx = -3.4e38f;
#pragma unroll
      for (int mm = 0; mm < 64; ++mm) mx = fmaxf(mx, lg[mm]);
      float se = 0.f;
#pragma unroll
      for (int mm = 0; mm < 64; ++mm){ float ev = __expf(lg[mm] - mx); lg[mm] = ev; se += ev; }
      float inv = 1.f / se;
      unsigned* wp = wbuf + (size_t)n * 128 + h * 32;
#pragma unroll
      for (int j = 0; j < 32; ++j) wp[j] = pack2(lg[2 * j] * inv, lg[2 * j + 1] * inv);
    }
  }
}

// ---------------- attention: z = sum_n w[n,h,m]*x[n,h,d] (chunked reduction) ----------------
__global__ __launch_bounds__(256) void k_attz(const unsigned* __restrict__ xln,
                                              const unsigned* __restrict__ wbuf,
                                              float* __restrict__ z, float* __restrict__ wsum, int N)
{
  __shared__ unsigned xs[64 * 64];     // 16KB
  __shared__ unsigned ws[64 * 128];    // 32KB
  int tid = threadIdx.x;
  int n0 = blockIdx.x * 64;
  for (int f = tid; f < 64 * 64; f += 256){
    int r = f >> 6; int n = n0 + r;
    xs[f] = (n < N) ? xln[(size_t)n * 64 + (f & 63)] : 0u;
  }
  for (int f = tid; f < 64 * 128; f += 256){
    int r = f >> 7; int n = n0 + r;
    ws[f] = (n < N) ? wbuf[(size_t)n * 128 + (f & 127)] : 0u;
  }
  __syncthreads();
  int h = tid >> 6, m = tid & 63;
  float acc[32];
#pragma unroll
  for (int d = 0; d < 32; ++d) acc[d] = 0.f;
  float wsacc = 0.f;
  const unsigned short* wsp = (const unsigned short*)ws;
  for (int n = 0; n < 64; ++n){
    float wv = bf2f(wsp[n * 256 + h * 64 + m]);
    wsacc += wv;
    const unsigned* xp = xs + n * 64 + h * 16;
#pragma unroll
    for (int j = 0; j < 16; ++j){
      unsigned u = xp[j];
      acc[2 * j]     += wv * bf2f((unsigned short)(u & 0xFFFFu));
      acc[2 * j + 1] += wv * bf2f((unsigned short)(u >> 16));
    }
  }
  int base = (h * 64 + m) * 32;
#pragma unroll
  for (int d = 0; d < 32; ++d) atomicAdd(&z[base + d], acc[d]);
  atomicAdd(&wsum[h * 64 + m], wsacc);
}

// ---------------- attention core (tiny, fp32, single block) ----------------
__global__ __launch_bounds__(256) void k_attcore(
  const float* __restrict__ z, const float* __restrict__ wsum,
  const float* __restrict__ Wq, const float* __restrict__ Wk, const float* __restrict__ Wv,
  float* __restrict__ obuf)
{
  __shared__ float zb[8192];
  __shared__ float wq[1024], wk[1024], wv[1024];
  int tid = threadIdx.x;
  for (int f = tid; f < 1024; f += 256){ wq[f] = Wq[f]; wk[f] = Wk[f]; wv[f] = Wv[f]; }
  int h = tid >> 6, m = tid & 63;
  int base = (h * 64 + m) * 32;
  float inv = 1.f / (wsum[h * 64 + m] + 1e-6f);
  float zr[32];
#pragma unroll
  for (int d = 0; d < 32; ++d){ zr[d] = z[base + d] * inv; zb[base + d] = zr[d]; }
  __syncthreads();
  float q[32];
  for (int d = 0; d < 32; ++d){
    float s = 0.f;
    for (int j = 0; j < 32; ++j) s += zr[j] * wq[j * 32 + d];
    q[d] = s;
  }
  float qk[32];
  for (int j = 0; j < 32; ++j){
    float s = 0.f;
    for (int d = 0; d < 32; ++d) s += q[d] * wk[j * 32 + d];
    qk[j] = s;
  }
  const float scale = 0.17677669529663687f;  // 1/sqrt(32)
  float lg[64]; float mx = -3.4e38f;
  for (int t = 0; t < 64; ++t){
    const float* zt = zb + (h * 64 + t) * 32;
    float s = 0.f;
    for (int j = 0; j < 32; ++j) s += zt[j] * qk[j];
    lg[t] = s * scale; mx = fmaxf(mx, lg[t]);
  }
  float se = 0.f;
  for (int t = 0; t < 64; ++t){ float ev = __expf(lg[t] - mx); lg[t] = ev; se += ev; }
  float si = 1.f / se;
  float pz[32];
#pragma unroll
  for (int j = 0; j < 32; ++j) pz[j] = 0.f;
  for (int t = 0; t < 64; ++t){
    float p = lg[t] * si;
    const float* zt = zb + (h * 64 + t) * 32;
#pragma unroll
    for (int j = 0; j < 32; ++j) pz[j] += p * zt[j];
  }
  for (int d = 0; d < 32; ++d){
    float s = 0.f;
    for (int j = 0; j < 32; ++j) s += pz[j] * wv[j * 32 + d];
    obuf[base + d] = s;
  }
}

// ---------------- de-slice: att[n,h,d] = sum_m w[n,h,m]*o[h,m,d] ----------------
__global__ __launch_bounds__(256) void k_deslice(const unsigned* __restrict__ wbuf,
                                                 const float* __restrict__ obuf,
                                                 unsigned* __restrict__ attb, int N)
{
  __shared__ float oL[8192];
  int tid = threadIdx.x;
  for (int f = tid; f < 8192; f += 256) oL[f] = obuf[f];
  __syncthreads();
  for (int n = blockIdx.x * 256 + tid; n < N; n += gridDim.x * 256){
    for (int h = 0; h < 4; ++h){
      float acc[32];
#pragma unroll
      for (int d = 0; d < 32; ++d) acc[d] = 0.f;
      const unsigned short* wp = (const unsigned short*)(wbuf + (size_t)n * 128) + h * 64;
      for (int mm = 0; mm < 64; ++mm){
        float wv = bf2f(wp[mm]);
        const float* op = oL + (h * 64 + mm) * 32;
#pragma unroll
        for (int d = 0; d < 32; ++d) acc[d] += wv * op[d];
      }
      unsigned* ap = attb + (size_t)n * 64 + h * 16;
#pragma unroll
      for (int j = 0; j < 16; ++j) ap[j] = pack2(acc[2 * j], acc[2 * j + 1]);
    }
  }
}

// ---------------- GEMM with += epilogue (Wo and FFN2) ----------------
__global__ __launch_bounds__(256) void k_gemm_add(const unsigned* __restrict__ A, int strideU, int KT,
                                                  const unsigned* __restrict__ Bs,
                                                  const float* __restrict__ bias,
                                                  float* __restrict__ outV, int N)
{
  int tid = threadIdx.x, wave = tid >> 5, lane = tid & 31;
  int n0 = blockIdx.x * 64;
  int mt = wave & 3, ntb = (wave >> 2) << 2;            // 4 m-tiles x 8 n-tiles, 4 nt/wave
  int arow = n0 + mt * 16 + (lane & 15); if (arow > N - 1) arow = N - 1;
  v8f acc[4];
#pragma unroll
  for (int j = 0; j < 4; ++j) acc[j] = vzero();
  for (int kt = 0; kt < KT; ++kt){
    Frag a; load_a(a, A, strideU, arow, kt, lane);
#pragma unroll
    for (int j = 0; j < 4; ++j){
      Frag b; load_b(b, Bs, 8, kt, ntb + j, lane);
      acc[j] = wmma_bf16(a, b, acc[j]);
    }
  }
  int hi = (lane >> 4) << 3;
#pragma unroll
  for (int j = 0; j < 4; ++j){
    int col = (ntb + j) * 16 + (lane & 15);
#pragma unroll
    for (int i = 0; i < 8; ++i){
      int row = n0 + mt * 16 + i + hi;
      if (row < N){
        size_t ix = (size_t)row * 128 + col;
        outV[ix] += acc[j][i] + bias[col];
      }
    }
  }
}

// ---------------- FFN1: hf = silu(xln @ Wf1 + bf1), 128->256 ----------------
__global__ __launch_bounds__(256) void k_ffn1(const unsigned* __restrict__ xln,
                                              const unsigned* __restrict__ Wf1s,
                                              const float* __restrict__ bf1,
                                              unsigned* __restrict__ hf, int N)
{
  int tid = threadIdx.x, wave = tid >> 5, lane = tid & 31;
  int n0 = blockIdx.x * 64;
  int mt = wave & 3, ntb = (wave >> 2) << 3;            // 4 m-tiles x 16 n-tiles, 8 nt/wave
  int arow = n0 + mt * 16 + (lane & 15); if (arow > N - 1) arow = N - 1;
  v8f acc[8];
#pragma unroll
  for (int j = 0; j < 8; ++j) acc[j] = vzero();
  for (int kt = 0; kt < 4; ++kt){
    Frag a; load_a(a, xln, 64, arow, kt, lane);
#pragma unroll
    for (int j = 0; j < 8; ++j){
      Frag b; load_b(b, Wf1s, 16, kt, ntb + j, lane);
      acc[j] = wmma_bf16(a, b, acc[j]);
    }
  }
  int hi = (lane >> 4) << 3;
#pragma unroll
  for (int j = 0; j < 8; ++j){
    int col = (ntb + j) * 16 + (lane & 15);
#pragma unroll
    for (int i = 0; i < 8; ++i){
      int row = n0 + mt * 16 + i + hi;
      if (row < N)
        ((unsigned short*)hf)[(size_t)row * 256 + col] = f2bf(siluf(acc[j][i] + bf1[col]));
    }
  }
}

// =====================================================================
extern "C" void kernel_launch(void* const* d_in, const int* in_sizes, int n_in,
                              void* d_out, int out_size, void* d_ws, size_t ws_size,
                              hipStream_t stream) {
  const float* V     = (const float*)d_in[0];
  const float* E     = (const float*)d_in[1];
  const float* s_enc = (const float*)d_in[2];
  const int*   edges = (const int*)d_in[3];
  const float* We1 = (const float*)d_in[4];  const float* be1 = (const float*)d_in[5];
  const float* We2 = (const float*)d_in[6];  const float* be2 = (const float*)d_in[7];
  const float* ge  = (const float*)d_in[8];  const float* beL = (const float*)d_in[9];
  const float* Wn1 = (const float*)d_in[10]; const float* bn1 = (const float*)d_in[11];
  const float* Wn2 = (const float*)d_in[12]; const float* bn2 = (const float*)d_in[13];
  const float* gn  = (const float*)d_in[14]; const float* bnL = (const float*)d_in[15];
  const float* g1  = (const float*)d_in[16]; const float* b1  = (const float*)d_in[17];
  const float* g2  = (const float*)d_in[18]; const float* b2  = (const float*)d_in[19];
  const float* Ws  = (const float*)d_in[20];
  const float* Wq  = (const float*)d_in[21];
  const float* Wk  = (const float*)d_in[22];
  const float* Wv  = (const float*)d_in[23];
  const float* Wo  = (const float*)d_in[24]; const float* bo  = (const float*)d_in[25];
  const float* Wf1 = (const float*)d_in[26]; const float* bf1 = (const float*)d_in[27];
  const float* Wf2 = (const float*)d_in[28]; const float* bf2 = (const float*)d_in[29];

  const int N  = in_sizes[0] / 128;
  const int NE = in_sizes[1] / 128;
  const int* snd = edges;
  const int* rcv = edges + NE;

  float* outV = (float*)d_out;
  float* outE = outV + (size_t)N * 128;

  // workspace carve-out
  size_t off = 0;
  auto alloc = [&](size_t bytes) -> void* {
    void* p = (char*)d_ws + off;
    off += (bytes + 255) & ~(size_t)255;
    return p;
  };
  unsigned* vin  = (unsigned*)alloc((size_t)N * 80 * 4);    // bf16 [N][160]
  float*    agg  = (float*)   alloc((size_t)N * 128 * 4);
  unsigned* xln  = (unsigned*)alloc((size_t)N * 64 * 4);    // bf16 [N][128] (reused for LN1/LN2)
  unsigned* wbuf = (unsigned*)alloc((size_t)N * 128 * 4);   // bf16 [N][4][64]; aliased as hf [N][256]
  unsigned* attb = (unsigned*)alloc((size_t)N * 64 * 4);    // bf16 [N][128]
  float*    zbuf = (float*)   alloc(8192 * 4);
  float*    wsum = (float*)   alloc(256 * 4);
  float*    obuf = (float*)   alloc(8192 * 4);
  unsigned* We1s = (unsigned*)alloc(448 * 128 * 2);
  unsigned* We2s = (unsigned*)alloc(128 * 128 * 2);
  unsigned* Wn1s = (unsigned*)alloc(288 * 128 * 2);
  unsigned* Wn2s = (unsigned*)alloc(128 * 128 * 2);
  unsigned* Wf1s = (unsigned*)alloc(128 * 256 * 2);
  unsigned* Wf2s = (unsigned*)alloc(256 * 128 * 2);
  unsigned* Wos  = (unsigned*)alloc(128 * 128 * 2);
  unsigned* hf   = wbuf;  // reuse (wbuf dead after deslice; hf written in FFN1)

  auto gsz = [](long total, int bs){ long g = (total + bs - 1) / bs; return (unsigned)(g < 1 ? 1 : g); };

  // weight swizzle + input packing
  k_swz<<<gsz(448 * 64, 256), 256, 0, stream>>>(We1, We1s, 448, 128);
  k_swz<<<gsz(128 * 64, 256), 256, 0, stream>>>(We2, We2s, 128, 128);
  k_swz<<<gsz(288 * 64, 256), 256, 0, stream>>>(Wn1, Wn1s, 288, 128);
  k_swz<<<gsz(128 * 64, 256), 256, 0, stream>>>(Wn2, Wn2s, 128, 128);
  k_swz<<<gsz(128 * 128, 256), 256, 0, stream>>>(Wf1, Wf1s, 128, 256);
  k_swz<<<gsz(256 * 64, 256), 256, 0, stream>>>(Wf2, Wf2s, 256, 128);
  k_swz<<<gsz(128 * 64, 256), 256, 0, stream>>>(Wo,  Wos,  128, 128);
  k_vin<<<gsz((long)N * 80, 256), 256, 0, stream>>>(V, s_enc, vin, N);
  k_zero<<<gsz((long)N * 128, 256), 256, 0, stream>>>(agg, (long)N * 128);
  k_zero<<<gsz(8192 + 256, 256), 256, 0, stream>>>(zbuf, 8192 + 256);  // zbuf + wsum contiguous

  // GNN
  k_edge<<<gsz(NE, 32), 256, 0, stream>>>(E, vin, snd, rcv, We1s, be1, We2s, be2, ge, beL,
                                          outE, agg, NE);
  k_node<<<gsz(N, 32), 256, 0, stream>>>(V, vin, agg, Wn1s, bn1, Wn2s, bn2, gn, bnL, outV, N);

  // attention
  k_ln<<<gsz(N, 256), 256, 0, stream>>>(outV, g1, b1, xln, N);
  k_attslice<<<gsz(N, 256), 256, 0, stream>>>(xln, Ws, wbuf, N);
  k_attz<<<gsz(N, 64), 256, 0, stream>>>(xln, wbuf, zbuf, wsum, N);
  k_attcore<<<1, 256, 0, stream>>>(zbuf, wsum, Wq, Wk, Wv, obuf);
  k_deslice<<<gsz(N, 256), 256, 0, stream>>>(wbuf, obuf, attb, N);
  k_gemm_add<<<gsz(N, 64), 256, 0, stream>>>(attb, 64, 4, Wos, bo, outV, N);

  // FFN
  k_ln<<<gsz(N, 256), 256, 0, stream>>>(outV, g2, b2, xln, N);
  k_ffn1<<<gsz(N, 64), 256, 0, stream>>>(xln, Wf1s, bf1, hf, N);
  k_gemm_add<<<gsz(N, 64), 256, 0, stream>>>(hf, 128, 8, Wf2s, bf2, outV, N);
}
